// OneRecDecoder_3229815407335
// MI455X (gfx1250) — compile-verified, matrix operand order checked by reading, not run on hardware
//
#include <hip/hip_runtime.h>
#include <math.h>

// ---------------------------------------------------------------------------
// OneRec decoder for MI455X (gfx1250): wave32 + WMMA f32_16x16x32_f16
// ---------------------------------------------------------------------------
#define DIMD 1024
#define NH   16
#define HD   64
#define NEXP 8
#define FF   4096
#define NL   2
#define NR   3
#define NV   8192
#define NB   256
#define NS   512
#define NT   4            // R+1 tokens per sequence
#define NTOK (NB*NT)      // 1024 decoder tokens
#define BENC (NB*NS)      // 131072 encoder tokens

typedef _Float16 half_t;
typedef __attribute__((ext_vector_type(16))) _Float16 v16h;
typedef __attribute__((ext_vector_type(8)))  _Float16 v8h;
typedef __attribute__((ext_vector_type(8)))  float    v8f;

__device__ __forceinline__ float gelu_exact(float x) {
    return 0.5f * x * (1.0f + erff(x * 0.70710678118654752f));
}

// ---------------------------------------------------------------------------
// Generic WMMA GEMM:  C[z] = act( A[z] @ W[z] + bias[z] )
// Block tile 128x64, 256 threads = 8 wave32 waves (4 M x 2 N); each wave
// computes a 32x32 patch = 4 x (16x16) WMMA tiles per K-step of 32.
// f32 -> f16 conversion in registers, packed ds_store_b128 into double-
// buffered LDS; global loads software-pipelined two tiles ahead.
// Requires: M%128==0, N%64==0, Kd%32==0 (all call sites comply).
// ---------------------------------------------------------------------------
#define BMT 128
#define BNT 64
#define LDSROW 40        // halves per LDS row (32 K + pad), 80B: 16B-aligned

__global__ __launch_bounds__(256) void k_gemm(
    const float* __restrict__ A, const float* __restrict__ W,
    const float* __restrict__ bias, void* __restrict__ Cout,
    int M, int N, int Kd, int lda, int ldc,
    long long sA, long long sW, long long sB, long long sC,
    int act, int out_f16)
{
    __shared__ __align__(16) half_t As[2][BMT * LDSROW];
    __shared__ __align__(16) half_t Bs[2][BNT * LDSROW];

    const int tid  = threadIdx.x;
    const int lane = tid & 31;
    const int w    = tid >> 5;       // wave 0..7
    const int wm   = w >> 1;         // 0..3 -> 32-row slab
    const int wn   = w & 1;          // 0..1 -> 32-col slab
    const int r    = lane & 15;
    const int hh   = lane >> 4;      // lane half selects K octet
    const int gm0  = blockIdx.y * BMT;
    const int gn0  = blockIdx.x * BNT;
    const int z    = blockIdx.z;

    A += (long long)z * sA;
    W += (long long)z * sW;
    if (bias) bias += (long long)z * sB;

    // staging work assignment
    const int arow = tid >> 1;           // 0..127
    const int akb  = (tid & 1) * 16;     // k sub-base: 0 | 16
    const int bn   = tid & 63;           // 0..63 (output col)
    const int bkb  = (tid >> 6) * 8;     // k sub-base: 0,8,16,24

    const float* aptr = A + (long long)(gm0 + arow) * lda + akb;
    const float* wptr = W + (long long)bkb * N + (gn0 + bn);

    float4 ra[4];
    float  rb[8];

    auto load_regs = [&](int k0) {
        const float4* ap = (const float4*)(aptr + k0);
#pragma unroll
        for (int j = 0; j < 4; ++j) ra[j] = ap[j];
        const float* wp = wptr + (long long)k0 * N;
#pragma unroll
        for (int j = 0; j < 8; ++j) rb[j] = wp[(long long)j * N];
    };

    auto store_lds = [&](int buf) {
        v8h pa0, pa1, pb;
#pragma unroll
        for (int j = 0; j < 4; ++j) {
            pa0[2 * j]     = (half_t)((j < 2 ? (j == 0 ? ra[0].x : ra[0].z)
                                             : (j == 2 ? ra[1].x : ra[1].z)));
            pa0[2 * j + 1] = (half_t)((j < 2 ? (j == 0 ? ra[0].y : ra[0].w)
                                             : (j == 2 ? ra[1].y : ra[1].w)));
            pa1[2 * j]     = (half_t)((j < 2 ? (j == 0 ? ra[2].x : ra[2].z)
                                             : (j == 2 ? ra[3].x : ra[3].z)));
            pa1[2 * j + 1] = (half_t)((j < 2 ? (j == 0 ? ra[2].y : ra[2].w)
                                             : (j == 2 ? ra[3].y : ra[3].w)));
        }
#pragma unroll
        for (int j = 0; j < 8; ++j) pb[j] = (half_t)rb[j];
        *(v8h*)&As[buf][arow * LDSROW + akb]     = pa0;
        *(v8h*)&As[buf][arow * LDSROW + akb + 8] = pa1;
        *(v8h*)&Bs[buf][bn * LDSROW + bkb]       = pb;
    };

    // ISA 7.12.2 fragment gather: 16-bit A 16x32, K pairs per (lane-half, j)
    auto load_frag = [&](const half_t* base, int rowbase) -> v16h {
        v16h f;
        const half_t* p = base + (rowbase + r) * LDSROW;
#pragma unroll
        for (int j = 0; j < 4; ++j) {
            int kk = hh * 8 + 2 * j;
            f[2 * j] = p[kk]; f[2 * j + 1] = p[kk + 1];
        }
#pragma unroll
        for (int j = 4; j < 8; ++j) {
            int kk = 16 + hh * 8 + 2 * (j - 4);
            f[2 * j] = p[kk]; f[2 * j + 1] = p[kk + 1];
        }
        return f;
    };

    v8f c00 = {}, c01 = {}, c10 = {}, c11 = {};

    const int nk = Kd >> 5;
    load_regs(0);
    store_lds(0);
    if (nk > 1) load_regs(32);

    for (int i = 0; i < nk; ++i) {
        __syncthreads();
        if (i + 1 < nk) {
            store_lds((i + 1) & 1);            // other buffer: safe, no barrier
            if (i + 2 < nk) load_regs((i + 2) * 32);
        }
        const half_t* ab = As[i & 1];
        const half_t* bbuf = Bs[i & 1];
        v16h a0 = load_frag(ab,   wm * 32);
        v16h a1 = load_frag(ab,   wm * 32 + 16);
        v16h b0 = load_frag(bbuf, wn * 32);
        v16h b1 = load_frag(bbuf, wn * 32 + 16);
        c00 = __builtin_amdgcn_wmma_f32_16x16x32_f16(false, a0, false, b0, (short)0, c00, false, false);
        c01 = __builtin_amdgcn_wmma_f32_16x16x32_f16(false, a0, false, b1, (short)0, c01, false, false);
        c10 = __builtin_amdgcn_wmma_f32_16x16x32_f16(false, a1, false, b0, (short)0, c10, false, false);
        c11 = __builtin_amdgcn_wmma_f32_16x16x32_f16(false, a1, false, b1, (short)0, c11, false, false);
    }

    // epilogue: C/D layout -> (row = v + 8*lane_half, col = lane&15)
#pragma unroll
    for (int v = 0; v < 8; ++v) {
        int row0 = gm0 + wm * 32 + v + 8 * hh;
        int row1 = row0 + 16;
        int col0 = gn0 + wn * 32 + r;
        int col1 = col0 + 16;
        float x00 = c00[v], x01 = c01[v], x10 = c10[v], x11 = c11[v];
        if (bias) {
            float bc0 = bias[col0], bc1 = bias[col1];
            x00 += bc0; x01 += bc1; x10 += bc0; x11 += bc1;
        }
        if (act == 1) {
            x00 = gelu_exact(x00); x01 = gelu_exact(x01);
            x10 = gelu_exact(x10); x11 = gelu_exact(x11);
        }
        if (out_f16) {
            half_t* C = (half_t*)Cout + (long long)z * sC;
            C[(long long)row0 * ldc + col0] = (half_t)x00;
            C[(long long)row0 * ldc + col1] = (half_t)x01;
            C[(long long)row1 * ldc + col0] = (half_t)x10;
            C[(long long)row1 * ldc + col1] = (half_t)x11;
        } else {
            float* C = (float*)Cout + (long long)z * sC;
            C[(long long)row0 * ldc + col0] = x00;
            C[(long long)row0 * ldc + col1] = x01;
            C[(long long)row1 * ldc + col0] = x10;
            C[(long long)row1 * ldc + col1] = x11;
        }
    }
}

// ---------------------------------------------------------------------------
// Embedding: x[b, t, :] = t==0 ? bos : emb_tables[t-1, ids[b, t-1], :]
// ---------------------------------------------------------------------------
__global__ __launch_bounds__(256) void k_embed(
    const int* __restrict__ ids, const float* __restrict__ embt,
    const float* __restrict__ bos, float* __restrict__ x)
{
    int tok = blockIdx.x;            // 0..NTOK-1
    int b = tok >> 2, t = tok & 3;
    const float* src;
    if (t == 0) src = bos;
    else {
        int rr = t - 1;
        int id = ids[b * NR + rr];
        src = embt + ((size_t)rr * NV + id) * DIMD;
    }
    float4 v = ((const float4*)src)[threadIdx.x];
    ((float4*)(x + (size_t)tok * DIMD))[threadIdx.x] = v;
}

// ---------------------------------------------------------------------------
// Self-attention (T=4, causal): one wave32 per (b, head); lane owns d, d+32.
// ---------------------------------------------------------------------------
__global__ __launch_bounds__(32) void k_self_attn(
    const float* __restrict__ Q, const float* __restrict__ K,
    const float* __restrict__ V, float* __restrict__ O)
{
    int b = blockIdx.x >> 4;
    int h = blockIdx.x & 15;
    int lane = threadIdx.x;

    float q[4][2], k[4][2], v[4][2];
#pragma unroll
    for (int t = 0; t < 4; ++t)
#pragma unroll
        for (int u = 0; u < 2; ++u) {
            int off = (b * 4 + t) * DIMD + h * HD + lane + 32 * u;
            q[t][u] = Q[off]; k[t][u] = K[off]; v[t][u] = V[off];
        }

    float s[4][4];
#pragma unroll
    for (int tq = 0; tq < 4; ++tq)
#pragma unroll
        for (int tk = 0; tk < 4; ++tk) {
            float p = q[tq][0] * k[tk][0] + q[tq][1] * k[tk][1];
#pragma unroll
            for (int o = 16; o > 0; o >>= 1) p += __shfl_xor(p, o, 32);
            s[tq][tk] = p * 0.125f;   // 1/sqrt(64)
        }

#pragma unroll
    for (int tq = 0; tq < 4; ++tq) {
        float m = -1e30f;
        for (int tk = 0; tk <= tq; ++tk) m = fmaxf(m, s[tq][tk]);
        float pr[4]; float ssum = 0.f;
        for (int tk = 0; tk <= tq; ++tk) { pr[tk] = expf(s[tq][tk] - m); ssum += pr[tk]; }
        float inv = 1.f / ssum;
        float o0 = 0.f, o1 = 0.f;
        for (int tk = 0; tk <= tq; ++tk) { o0 += pr[tk] * v[tk][0]; o1 += pr[tk] * v[tk][1]; }
        int off = (b * 4 + tq) * DIMD + h * HD + lane;
        O[off] = o0 * inv; O[off + 32] = o1 * inv;
    }
}

// ---------------------------------------------------------------------------
// Cross-attention: block per (b, head); S=512 keys, K/V in f16.
// ---------------------------------------------------------------------------
__global__ __launch_bounds__(256) void k_cross_attn(
    const float* __restrict__ Q, const half_t* __restrict__ Ke,
    const half_t* __restrict__ Ve, float* __restrict__ O)
{
    __shared__ float qs[4][HD];
    __shared__ float p[4][NS];
    __shared__ float red[256];

    int b = blockIdx.x >> 4;
    int h = blockIdx.x & 15;
    int tid = threadIdx.x;

    {
        int tq = tid >> 6, d = tid & 63;
        qs[tq][d] = Q[(b * 4 + tq) * DIMD + h * HD + d] * 0.125f;
    }
    __syncthreads();

    // scores: K rows read as v8h chunks
    for (int s = tid; s < NS; s += 256) {
        const v8h* kr8 = (const v8h*)(Ke + ((long long)(b * NS + s)) * DIMD + h * HD);
        float acc0 = 0.f, acc1 = 0.f, acc2 = 0.f, acc3 = 0.f;
#pragma unroll
        for (int c = 0; c < 8; ++c) {
            v8h kc = kr8[c];
#pragma unroll
            for (int u = 0; u < 8; ++u) {
                float kv = (float)kc[u];
                int d = c * 8 + u;
                acc0 += qs[0][d] * kv; acc1 += qs[1][d] * kv;
                acc2 += qs[2][d] * kv; acc3 += qs[3][d] * kv;
            }
        }
        p[0][s] = acc0; p[1][s] = acc1; p[2][s] = acc2; p[3][s] = acc3;
    }
    __syncthreads();

    // softmax over S per query
    for (int tq = 0; tq < 4; ++tq) {
        red[tid] = fmaxf(p[tq][tid], p[tq][tid + 256]);
        __syncthreads();
        for (int o = 128; o > 0; o >>= 1) {
            if (tid < o) red[tid] = fmaxf(red[tid], red[tid + o]);
            __syncthreads();
        }
        float m = red[0];
        __syncthreads();
        float e0 = expf(p[tq][tid] - m), e1 = expf(p[tq][tid + 256] - m);
        p[tq][tid] = e0; p[tq][tid + 256] = e1;
        red[tid] = e0 + e1;
        __syncthreads();
        for (int o = 128; o > 0; o >>= 1) {
            if (tid < o) red[tid] += red[tid + o];
            __syncthreads();
        }
        float inv = 1.f / red[0];
        __syncthreads();
        p[tq][tid] *= inv; p[tq][tid + 256] *= inv;
        __syncthreads();
    }

    // O[tq, d] = sum_s p[tq][s] * V[s, d]   (coalesced across lanes in d)
    {
        int tq = tid >> 6, d = tid & 63;
        const half_t* vc = Ve + (long long)(b * NS) * DIMD + h * HD + d;
        float acc = 0.f;
#pragma unroll 4
        for (int s = 0; s < NS; ++s) acc += p[tq][s] * (float)vc[(long long)s * DIMD];
        O[(b * 4 + tq) * DIMD + h * HD + d] = acc;
    }
}

// ---------------------------------------------------------------------------
// out = LayerNorm(x + res) * g + b   (res may be null). Block per token.
// ---------------------------------------------------------------------------
__global__ __launch_bounds__(256) void k_add_ln(
    const float* __restrict__ x, const float* __restrict__ res,
    const float* __restrict__ g, const float* __restrict__ bb,
    float* __restrict__ out)
{
    __shared__ float r1[256], r2[256];
    int t = blockIdx.x, tid = threadIdx.x;
    const float* xp = x + (size_t)t * DIMD;
    const float* rp = res ? res + (size_t)t * DIMD : nullptr;

    float vals[4]; float s = 0.f, sq = 0.f;
#pragma unroll
    for (int i = 0; i < 4; ++i) {
        int d = tid + 256 * i;
        float v = xp[d] + (rp ? rp[d] : 0.f);
        vals[i] = v; s += v; sq += v * v;
    }
    r1[tid] = s; r2[tid] = sq;
    __syncthreads();
    for (int o = 128; o > 0; o >>= 1) {
        if (tid < o) { r1[tid] += r1[tid + o]; r2[tid] += r2[tid + o]; }
        __syncthreads();
    }
    float mean = r1[0] * (1.f / DIMD);
    float var  = r2[0] * (1.f / DIMD) - mean * mean;
    float inv  = rsqrtf(var + 1e-5f);
#pragma unroll
    for (int i = 0; i < 4; ++i) {
        int d = tid + 256 * i;
        out[(size_t)t * DIMD + d] = (vals[i] - mean) * inv * g[d] + bb[d];
    }
}

// ---------------------------------------------------------------------------
// Router + top-2 gating (E=8). Block per token.
// ---------------------------------------------------------------------------
__global__ __launch_bounds__(256) void k_router(
    const float* __restrict__ x, const float* __restrict__ rw,
    const float* __restrict__ rb, float* __restrict__ gate)
{
    __shared__ float red[256];
    __shared__ float lg[NEXP];
    int t = blockIdx.x, tid = threadIdx.x;
    const float* xp = x + (size_t)t * DIMD;

    float part[NEXP] = {0, 0, 0, 0, 0, 0, 0, 0};
    for (int d = tid; d < DIMD; d += 256) {
        float xv = xp[d];
#pragma unroll
        for (int e = 0; e < NEXP; ++e) part[e] += xv * rw[d * NEXP + e];
    }
    for (int e = 0; e < NEXP; ++e) {
        red[tid] = part[e];
        __syncthreads();
        for (int o = 128; o > 0; o >>= 1) {
            if (tid < o) red[tid] += red[tid + o];
            __syncthreads();
        }
        if (tid == 0) lg[e] = red[0] + rb[e];
        __syncthreads();
    }
    if (tid == 0) {
        float m = lg[0];
        for (int e = 1; e < NEXP; ++e) m = fmaxf(m, lg[e]);
        float pr[NEXP]; float ssum = 0.f;
        for (int e = 0; e < NEXP; ++e) { pr[e] = expf(lg[e] - m); ssum += pr[e]; }
        for (int e = 0; e < NEXP; ++e) pr[e] /= ssum;
        int i1 = 0;
        for (int e = 1; e < NEXP; ++e) if (pr[e] > pr[i1]) i1 = e;
        int i2 = (i1 == 0) ? 1 : 0;
        for (int e = 0; e < NEXP; ++e) if (e != i1 && pr[e] > pr[i2]) i2 = e;
        float e1 = expf(pr[i1]), e2 = expf(pr[i2]);
        float den = 1.f / (e1 + e2);
        for (int e = 0; e < NEXP; ++e) gate[t * NEXP + e] = 0.f;
        gate[t * NEXP + i1] = e1 * den;
        gate[t * NEXP + i2] = e2 * den;
    }
}

// ---------------------------------------------------------------------------
// MoE combine: tmp[t, d] = sum_e gate[t, e] * y[e, t, d]   (float4 per thread)
// ---------------------------------------------------------------------------
__global__ __launch_bounds__(256) void k_moe_combine(
    const float* __restrict__ y, const float* __restrict__ gate,
    float* __restrict__ tmp)
{
    int idx4 = blockIdx.x * 256 + threadIdx.x;   // float4 index, < NTOK*DIMD/4
    int t = idx4 >> 8;                           // (idx4*4)>>10
    float a0 = 0.f, a1 = 0.f, a2 = 0.f, a3 = 0.f;
#pragma unroll
    for (int e = 0; e < NEXP; ++e) {
        float ge = gate[t * NEXP + e];
        float4 v = ((const float4*)y)[((size_t)e << 18) + idx4];
        a0 += ge * v.x; a1 += ge * v.y; a2 += ge * v.z; a3 += ge * v.w;
    }
    float4 o; o.x = a0; o.y = a1; o.z = a2; o.w = a3;
    ((float4*)tmp)[idx4] = o;
}

// ---------------------------------------------------------------------------
// Host-side orchestration
// ---------------------------------------------------------------------------
static inline void gemm(hipStream_t stream, const float* A, const float* W,
                        const float* bias, void* C,
                        int M, int N, int Kd, int lda, int ldc,
                        long long sA, long long sW, long long sB, long long sC,
                        int batch, int act, int out_f16)
{
    dim3 g(N / BNT, M / BMT, batch);
    k_gemm<<<g, 256, 0, stream>>>(A, W, bias, C, M, N, Kd, lda, ldc,
                                  sA, sW, sB, sC, act, out_f16);
}

extern "C" void kernel_launch(void* const* d_in, const int* in_sizes, int n_in,
                              void* d_out, int out_size, void* d_ws, size_t ws_size,
                              hipStream_t stream) {
    (void)in_sizes; (void)n_in; (void)out_size; (void)ws_size;
    const int*   ids    = (const int*)  d_in[0];
    const float* enc    = (const float*)d_in[1];
    const float* embt   = (const float*)d_in[2];
    const float* bos    = (const float*)d_in[3];
    const float* sa_wq  = (const float*)d_in[4];
    const float* sa_bq  = (const float*)d_in[5];
    const float* sa_wk  = (const float*)d_in[6];
    const float* sa_bk  = (const float*)d_in[7];
    const float* sa_wv  = (const float*)d_in[8];
    const float* sa_bv  = (const float*)d_in[9];
    const float* sa_wo  = (const float*)d_in[10];
    const float* sa_bo  = (const float*)d_in[11];
    const float* ca_wq  = (const float*)d_in[12];
    const float* ca_bq  = (const float*)d_in[13];
    const float* ca_wk  = (const float*)d_in[14];
    const float* ca_bk  = (const float*)d_in[15];
    const float* ca_wv  = (const float*)d_in[16];
    const float* ca_bv  = (const float*)d_in[17];
    const float* ca_wo  = (const float*)d_in[18];
    const float* ca_bo  = (const float*)d_in[19];
    const float* rt_w   = (const float*)d_in[20];
    const float* rt_b   = (const float*)d_in[21];
    const float* moe_w1 = (const float*)d_in[22];
    const float* moe_b1 = (const float*)d_in[23];
    const float* moe_w2 = (const float*)d_in[24];
    const float* moe_b2 = (const float*)d_in[25];
    const float* ln1_g  = (const float*)d_in[26];
    const float* ln1_b  = (const float*)d_in[27];
    const float* ln2_g  = (const float*)d_in[28];
    const float* ln2_b  = (const float*)d_in[29];
    const float* ln3_g  = (const float*)d_in[30];
    const float* ln3_b  = (const float*)d_in[31];
    const float* lnf_g  = (const float*)d_in[32];
    const float* lnf_b  = (const float*)d_in[33];
    const float* out_w  = (const float*)d_in[34];
    const float* out_b  = (const float*)d_in[35];
    float* out = (float*)d_out;

    char* wp = (char*)d_ws;
    auto alloc = [&](size_t bytes) -> void* {
        void* rr = (void*)wp;
        wp += (bytes + 255) & ~(size_t)255;
        return rr;
    };
    float*  x    = (float*) alloc((size_t)NTOK * DIMD * 4);
    float*  qb   = (float*) alloc((size_t)NTOK * DIMD * 4);
    float*  kb   = (float*) alloc((size_t)NTOK * DIMD * 4);
    float*  vb   = (float*) alloc((size_t)NTOK * DIMD * 4);
    float*  ao   = (float*) alloc((size_t)NTOK * DIMD * 4);
    float*  tmp  = (float*) alloc((size_t)NTOK * DIMD * 4);
    half_t* kenc = (half_t*)alloc((size_t)BENC * DIMD * 2);
    half_t* venc = (half_t*)alloc((size_t)BENC * DIMD * 2);
    float*  gate = (float*) alloc((size_t)NTOK * NEXP * 4);
    float*  hbuf = (float*) alloc((size_t)NEXP * NTOK * FF * 4);
    float*  ybuf = (float*) alloc((size_t)NEXP * NTOK * DIMD * 4);

    const size_t Dsq = (size_t)DIMD * DIMD;

    k_embed<<<NTOK, 256, 0, stream>>>(ids, embt, bos, x);

    for (int i = 0; i < NL; ++i) {
        // ---- self-attention ----
        gemm(stream, x, sa_wq + i * Dsq, sa_bq + i * DIMD, qb,
             NTOK, DIMD, DIMD, DIMD, DIMD, 0, 0, 0, 0, 1, 0, 0);
        gemm(stream, x, sa_wk + i * Dsq, sa_bk + i * DIMD, kb,
             NTOK, DIMD, DIMD, DIMD, DIMD, 0, 0, 0, 0, 1, 0, 0);
        gemm(stream, x, sa_wv + i * Dsq, sa_bv + i * DIMD, vb,
             NTOK, DIMD, DIMD, DIMD, DIMD, 0, 0, 0, 0, 1, 0, 0);
        k_self_attn<<<NB * NH, 32, 0, stream>>>(qb, kb, vb, ao);
        gemm(stream, ao, sa_wo + i * Dsq, sa_bo + i * DIMD, tmp,
             NTOK, DIMD, DIMD, DIMD, DIMD, 0, 0, 0, 0, 1, 0, 0);
        k_add_ln<<<NTOK, 256, 0, stream>>>(x, tmp, ln1_g + i * DIMD, ln1_b + i * DIMD, x);

        // ---- cross-attention ----
        gemm(stream, x, ca_wq + i * Dsq, ca_bq + i * DIMD, qb,
             NTOK, DIMD, DIMD, DIMD, DIMD, 0, 0, 0, 0, 1, 0, 0);
        gemm(stream, enc, ca_wk + i * Dsq, ca_bk + i * DIMD, kenc,
             BENC, DIMD, DIMD, DIMD, DIMD, 0, 0, 0, 0, 1, 0, 1);
        gemm(stream, enc, ca_wv + i * Dsq, ca_bv + i * DIMD, venc,
             BENC, DIMD, DIMD, DIMD, DIMD, 0, 0, 0, 0, 1, 0, 1);
        k_cross_attn<<<NB * NH, 256, 0, stream>>>(qb, kenc, venc, ao);
        gemm(stream, ao, ca_wo + i * Dsq, ca_bo + i * DIMD, tmp,
             NTOK, DIMD, DIMD, DIMD, DIMD, 0, 0, 0, 0, 1, 0, 0);
        k_add_ln<<<NTOK, 256, 0, stream>>>(x, tmp, ln2_g + i * DIMD, ln2_b + i * DIMD, x);

        // ---- MoE (dense per-expert, batched over grid.z) ----
        k_router<<<NTOK, 256, 0, stream>>>(x, rt_w + (size_t)i * DIMD * NEXP,
                                           rt_b + i * NEXP, gate);
        gemm(stream, x, moe_w1 + (size_t)i * NEXP * DIMD * FF,
             moe_b1 + (size_t)i * NEXP * FF, hbuf,
             NTOK, FF, DIMD, DIMD, FF,
             0, (long long)DIMD * FF, FF, (long long)NTOK * FF,
             NEXP, /*gelu*/1, 0);
        gemm(stream, hbuf, moe_w2 + (size_t)i * NEXP * FF * DIMD,
             moe_b2 + (size_t)i * NEXP * DIMD, ybuf,
             NTOK, DIMD, FF, FF, DIMD,
             (long long)NTOK * FF, (long long)FF * DIMD, DIMD,
             (long long)NTOK * DIMD, NEXP, 0, 0);
        k_moe_combine<<<(NTOK * DIMD / 4) / 256, 256, 0, stream>>>(ybuf, gate, tmp);
        k_add_ln<<<NTOK, 256, 0, stream>>>(x, tmp, ln3_g + i * DIMD, ln3_b + i * DIMD, x);
    }

    k_add_ln<<<NTOK, 256, 0, stream>>>(x, nullptr, lnf_g, lnf_b, x);

    // logits: out[:, r, :] = x[:, r, :] @ out_w[r] + out_b[r]
    gemm(stream, x, out_w, out_b, out,
         NB, NV, DIMD, NT * DIMD, NR * NV,
         DIMD, (long long)DIMD * NV, NV, NV,
         NR, 0, 0);
}
